// KuaiLstm_5823975654031
// MI455X (gfx1250) — compile-verified, matrix operand order checked by reading, not run on hardware
//
#include <hip/hip_runtime.h>
#include <hip/hip_bf16.h>
#include <math.h>

// ---------------- problem constants ----------------
#define NT      365
#define NGRID   2048
#define NX      32
#define HID     256
#define NG      1024          // 4*HID gates
#define KTOT    512           // K = [x0(256) | h(256)]
#define TM      16            // batch rows per workgroup
#define NTHREADS 512          // 16 waves
#define NWG     (NGRID / TM)  // 128 workgroups

// fragment counts for repacked weights
#define WCAT_FRAGS (64 * 16)        // 64 N-tiles x 16 K-steps
#define WCAT_ELEMS (WCAT_FRAGS * 512)
#define WIN_FRAGS  16               // 16 N-tiles x 1 K-step
#define WIN_ELEMS  (WIN_FRAGS * 512)

typedef __attribute__((ext_vector_type(16))) __bf16 v16bf;
typedef __attribute__((ext_vector_type(8)))  float  v8f;

union BFrag {
    unsigned int u[8];
    v16bf        v;
};

__device__ __forceinline__ unsigned short f2bf(float f) {
    union { __bf16 b; unsigned short s; } u;
    u.b = (__bf16)f;
    return u.s;
}

__device__ __forceinline__ unsigned int pk2bf(float lo, float hi) {
    return (unsigned int)f2bf(lo) | ((unsigned int)f2bf(hi) << 16);
}

__device__ __forceinline__ float fsigmoid(float x) {
    return 1.0f / (1.0f + __expf(-x));
}

__device__ __forceinline__ float bflo(unsigned int u) {
    return __uint_as_float(u << 16);
}
__device__ __forceinline__ float bfhi(unsigned int u) {
    return __uint_as_float(u & 0xffff0000u);
}

// ---------------------------------------------------------------------------
// Prep: repack W_in^T and [W_ih;W_hh]^T (f32) into bf16 WMMA B-fragment order.
// B fragment (32x16 bf16): lane L holds column N = L%16; VGPR v, half h holds
// K = (v&3)*2 + h + (v>>2)*16 + (L>>4)*8.  Per-frag storage: 32 lanes x 16
// consecutive bf16 -> hot loop reads two uint4 per lane, fully coalesced.
// ---------------------------------------------------------------------------
__global__ void prep_weights(const float* __restrict__ w_in,
                             const float* __restrict__ w_ih,
                             const float* __restrict__ w_hh,
                             unsigned short* __restrict__ wcat_frag,
                             unsigned short* __restrict__ win_frag) {
    int idx = blockIdx.x * blockDim.x + threadIdx.x;
    if (idx < WCAT_ELEMS) {
        int frag   = idx >> 9;        // kk*64 + nt
        int within = idx & 511;
        int lane = within >> 4;
        int e    = within & 15;
        int v = e >> 1, h = e & 1;
        int kk = frag >> 6;
        int nt = frag & 63;
        int K = kk * 32 + (v & 3) * 2 + h + (v >> 2) * 16 + (lane >> 4) * 8;
        int N = nt * 16 + (lane & 15);
        float val = (K < HID) ? w_ih[N * HID + K] : w_hh[N * HID + (K - HID)];
        wcat_frag[idx] = f2bf(val);
    } else {
        int j = idx - WCAT_ELEMS;
        if (j < WIN_ELEMS) {
            int nt     = j >> 9;
            int within = j & 511;
            int lane = within >> 4;
            int e    = within & 15;
            int v = e >> 1, h = e & 1;
            int K = (v & 3) * 2 + h + (v >> 2) * 16 + (lane >> 4) * 8; // 0..31
            int N = nt * 16 + (lane & 15);
            win_frag[j] = f2bf(w_in[N * NX + K]);
        }
    }
}

// ---------------------------------------------------------------------------
// Persistent fused LSTM. One workgroup owns TM=16 batch rows for all NT steps.
// 16 waves: wave wn (0..15) owns hidden-column tile wn for ALL FOUR gates
// (4 accumulator tiles), so the cell update runs straight on the WMMA
// accumulators; cell state c stays in registers across the whole time loop.
// K-loop uses a manual ping-pong double buffer of B fragments so the 8 global
// B-loads of step k+1 overlap the 4 WMMAs of step k.
// ---------------------------------------------------------------------------
__global__ void __launch_bounds__(NTHREADS, 1)
lstm_persist(const float* __restrict__ x,
             const float* __restrict__ b_in,
             const float* __restrict__ b_ih,
             const float* __restrict__ b_hh,
             const float* __restrict__ w_out,
             const float* __restrict__ b_out,
             const unsigned short* __restrict__ wcat_frag_u,
             const unsigned short* __restrict__ win_frag_u,
             float* __restrict__ out) {
    __shared__ unsigned short abuf[TM * KTOT];  // 16KB: [row][512] bf16, x0|h
    __shared__ float s_bias[NG];                // b_ih + b_hh
    __shared__ float s_bin[HID];
    __shared__ float s_wout[HID];

    const int tid    = threadIdx.x;
    const int wn     = tid >> 5;      // wave 0..15 -> hidden column tile
    const int lane   = tid & 31;
    const int laneHi = lane >> 4;
    const int nLo    = lane & 15;
    const int batch0 = blockIdx.x * TM;

    // small vectors -> LDS (once)
    for (int i = tid; i < NG; i += NTHREADS) s_bias[i] = b_ih[i] + b_hh[i];
    for (int i = tid; i < HID; i += NTHREADS) { s_bin[i] = b_in[i]; s_wout[i] = w_out[i]; }
    // h(t=0) = 0
    for (int i = tid; i < TM * HID; i += NTHREADS) {
        int r = i >> 8, cc = i & 255;
        abuf[r * KTOT + HID + cc] = 0;  // bf16 +0.0
    }
    const float bo = b_out[0];

    // persistent cell state: r -> row laneHi*8 + r, col j = wn*16 + nLo
    float c_st[8];
#pragma unroll
    for (int r = 0; r < 8; ++r) c_st[r] = 0.0f;

    const unsigned short* wc_lane = wcat_frag_u + lane * 16;

    __syncthreads();

#pragma unroll 1
    for (int t = 0; t < NT; ++t) {
        // ---- phase 1: x0 = relu(x[t]·W_in^T + b_in) -> abuf cols [0,256)
        {
            const float* xr =
                x + ((size_t)t * NGRID + batch0 + nLo) * NX + laneHi * 8;
            float4 f0 = *(const float4*)(xr + 0);
            float4 f1 = *(const float4*)(xr + 4);
            float4 f2 = *(const float4*)(xr + 16);
            float4 f3 = *(const float4*)(xr + 20);
            BFrag a;
            a.u[0] = pk2bf(f0.x, f0.y); a.u[1] = pk2bf(f0.z, f0.w);
            a.u[2] = pk2bf(f1.x, f1.y); a.u[3] = pk2bf(f1.z, f1.w);
            a.u[4] = pk2bf(f2.x, f2.y); a.u[5] = pk2bf(f2.z, f2.w);
            a.u[6] = pk2bf(f3.x, f3.y); a.u[7] = pk2bf(f3.z, f3.w);

            const uint4* bp = (const uint4*)(win_frag_u + wn * 512 + lane * 16);
            uint4 l4 = bp[0], h4 = bp[1];
            BFrag b;
            b.u[0] = l4.x; b.u[1] = l4.y; b.u[2] = l4.z; b.u[3] = l4.w;
            b.u[4] = h4.x; b.u[5] = h4.y; b.u[6] = h4.z; b.u[7] = h4.w;

            v8f c = {0.f, 0.f, 0.f, 0.f, 0.f, 0.f, 0.f, 0.f};
            c = __builtin_amdgcn_wmma_f32_16x16x32_bf16(
                    false, a.v, false, b.v, (short)0, c, false, false);
            int col = wn * 16 + nLo;
            float bi = s_bin[col];
#pragma unroll
            for (int r = 0; r < 8; ++r) {
                int row = laneHi * 8 + r;
                float val = c[r] + bi;
                val = val > 0.0f ? val : 0.0f;
                abuf[row * KTOT + col] = f2bf(val);
            }
        }
        __syncthreads();

        // ---- phase 2: gates = [x0|h]·[W_ih;W_hh]^T  (K=512, 16 K-steps)
        //      manual ping-pong double buffer on B fragments
        v8f acc[4];
#pragma unroll
        for (int g = 0; g < 4; ++g)
            acc[g] = (v8f){0.f, 0.f, 0.f, 0.f, 0.f, 0.f, 0.f, 0.f};

        BFrag b0[4], b1[4];
#pragma unroll
        for (int g = 0; g < 4; ++g) {   // preload kk = 0
            const uint4* bp =
                (const uint4*)(wc_lane + (size_t)(g * 16 + wn) * 512);
            uint4 l4 = bp[0], h4 = bp[1];
            b0[g].u[0] = l4.x; b0[g].u[1] = l4.y; b0[g].u[2] = l4.z; b0[g].u[3] = l4.w;
            b0[g].u[4] = h4.x; b0[g].u[5] = h4.y; b0[g].u[6] = h4.z; b0[g].u[7] = h4.w;
        }

#pragma unroll 1
        for (int kk = 0; kk < 16; kk += 2) {
            // ---- half A: uses b0 (kk), prefetches b1 (kk+1)
            const unsigned short* ar =
                abuf + nLo * KTOT + kk * 32 + laneHi * 8;
            uint4 a00 = *(const uint4*)(ar);
            uint4 a01 = *(const uint4*)(ar + 16);
            BFrag aA;
            aA.u[0] = a00.x; aA.u[1] = a00.y; aA.u[2] = a00.z; aA.u[3] = a00.w;
            aA.u[4] = a01.x; aA.u[5] = a01.y; aA.u[6] = a01.z; aA.u[7] = a01.w;

#pragma unroll
            for (int g = 0; g < 4; ++g) {
                const uint4* bp = (const uint4*)(wc_lane +
                    (size_t)(kk + 1) * (64 * 512) + (size_t)(g * 16 + wn) * 512);
                uint4 l4 = bp[0], h4 = bp[1];
                b1[g].u[0] = l4.x; b1[g].u[1] = l4.y; b1[g].u[2] = l4.z; b1[g].u[3] = l4.w;
                b1[g].u[4] = h4.x; b1[g].u[5] = h4.y; b1[g].u[6] = h4.z; b1[g].u[7] = h4.w;
            }
#pragma unroll
            for (int g = 0; g < 4; ++g)
                acc[g] = __builtin_amdgcn_wmma_f32_16x16x32_bf16(
                    false, aA.v, false, b0[g].v, (short)0, acc[g], false, false);

            // ---- half B: uses b1 (kk+1), prefetches b0 (kk+2, wrap-safe)
            uint4 a10 = *(const uint4*)(ar + 32);
            uint4 a11 = *(const uint4*)(ar + 48);
            BFrag aB;
            aB.u[0] = a10.x; aB.u[1] = a10.y; aB.u[2] = a10.z; aB.u[3] = a10.w;
            aB.u[4] = a11.x; aB.u[5] = a11.y; aB.u[6] = a11.z; aB.u[7] = a11.w;

            int kpre = (kk + 2) & 15;   // last prefetch wraps to valid memory
#pragma unroll
            for (int g = 0; g < 4; ++g) {
                const uint4* bp = (const uint4*)(wc_lane +
                    (size_t)kpre * (64 * 512) + (size_t)(g * 16 + wn) * 512);
                uint4 l4 = bp[0], h4 = bp[1];
                b0[g].u[0] = l4.x; b0[g].u[1] = l4.y; b0[g].u[2] = l4.z; b0[g].u[3] = l4.w;
                b0[g].u[4] = h4.x; b0[g].u[5] = h4.y; b0[g].u[6] = h4.z; b0[g].u[7] = h4.w;
            }
#pragma unroll
            for (int g = 0; g < 4; ++g)
                acc[g] = __builtin_amdgcn_wmma_f32_16x16x32_bf16(
                    false, aB.v, false, b1[g].v, (short)0, acc[g], false, false);
        }
        // all reads of abuf (x0|h) are done before h is overwritten
        __syncthreads();

        // ---- phase 3: LSTM cell directly on accumulators
        {
            int j = wn * 16 + nLo;
            float bi_ = s_bias[j];
            float bf_ = s_bias[HID + j];
            float bg_ = s_bias[2 * HID + j];
            float bo_ = s_bias[3 * HID + j];
#pragma unroll
            for (int r = 0; r < 8; ++r) {
                int row = laneHi * 8 + r;
                float gi = acc[0][r] + bi_;
                float gf = acc[1][r] + bf_;
                float gg = acc[2][r] + bg_;
                float go = acc[3][r] + bo_;
                float cn = fsigmoid(gf) * c_st[r] + fsigmoid(gi) * tanhf(gg);
                float hn = fsigmoid(go) * tanhf(cn);
                c_st[r] = cn;
                abuf[row * KTOT + HID + j] = f2bf(hn);  // h for next step
            }
        }
        __syncthreads();

        // ---- phase 4: out[t,b] = h·w_out + b_out ; wave wn reduces row wn
        {
            const uint4* hp = (const uint4*)(abuf + wn * KTOT + HID + lane * 8);
            uint4 hv = *hp;                         // 8 bf16 of h[row=wn]
            int cb = lane * 8;
            float s;
            s  = bflo(hv.x) * s_wout[cb + 0] + bfhi(hv.x) * s_wout[cb + 1];
            s += bflo(hv.y) * s_wout[cb + 2] + bfhi(hv.y) * s_wout[cb + 3];
            s += bflo(hv.z) * s_wout[cb + 4] + bfhi(hv.z) * s_wout[cb + 5];
            s += bflo(hv.w) * s_wout[cb + 6] + bfhi(hv.w) * s_wout[cb + 7];
#pragma unroll
            for (int off = 16; off >= 1; off >>= 1)
                s += __shfl_xor(s, off, 32);
            if (lane == 0)
                out[(size_t)t * NGRID + batch0 + wn] = s + bo;
        }
        // phase-1 of t+1 writes only x0 columns; next barrier orders the rest
    }
}

// ---------------------------------------------------------------------------
extern "C" void kernel_launch(void* const* d_in, const int* in_sizes, int n_in,
                              void* d_out, int out_size, void* d_ws,
                              size_t ws_size, hipStream_t stream) {
    const float* x     = (const float*)d_in[0];
    const float* w_in  = (const float*)d_in[1];
    const float* b_in  = (const float*)d_in[2];
    const float* w_ih  = (const float*)d_in[3];
    const float* w_hh  = (const float*)d_in[4];
    const float* b_ih  = (const float*)d_in[5];
    const float* b_hh  = (const float*)d_in[6];
    const float* w_out = (const float*)d_in[7];
    const float* b_out = (const float*)d_in[8];
    float* out = (float*)d_out;

    unsigned short* wcat_frag = (unsigned short*)d_ws;            // 1 MB
    unsigned short* win_frag  = wcat_frag + WCAT_ELEMS;           // 16 KB

    int total = WCAT_ELEMS + WIN_ELEMS;
    prep_weights<<<(total + 255) / 256, 256, 0, stream>>>(
        w_in, w_ih, w_hh, wcat_frag, win_frag);

    lstm_persist<<<NWG, NTHREADS, 0, stream>>>(
        x, b_in, b_ih, b_hh, w_out, b_out, wcat_frag, win_frag, out);
}
